// GroupedQueryAttention_81097572483597
// MI455X (gfx1250) — matrix operand order repaired and checked by reading, not compiled, not run on hardware
//
#include <hip/hip_runtime.h>
#include <math.h>
#include <stdint.h>

// ---------- types ----------
typedef __attribute__((ext_vector_type(16))) _Float16 v16h;
typedef __attribute__((ext_vector_type(8)))  _Float16 v8h;
typedef __attribute__((ext_vector_type(4)))  _Float16 v4h;
typedef __attribute__((ext_vector_type(8)))  float    v8f;
typedef __attribute__((ext_vector_type(4)))  unsigned int v4u;
typedef __attribute__((ext_vector_type(8)))  int v8i;
typedef __attribute__((ext_vector_type(4)))  int v4i;

// ---------- problem constants (fixed by reference) ----------
#define T_SEQ   2048
#define C_DIM   4096
#define KVD     2048
#define HD      128
#define NHEADS  32
#define NKV     16
#define SCALE_F 0.17677669529663687f   // 1/sqrt(32)

__device__ __forceinline__ v8f zero_v8f() {
  v8f z;
#pragma unroll
  for (int i = 0; i < 8; ++i) z[i] = 0.0f;
  return z;
}

__device__ __forceinline__ v16h cat8(v8h lo, v8h hi) {
  return __builtin_shufflevector(lo, hi, 0,1,2,3,4,5,6,7,8,9,10,11,12,13,14,15);
}

// CDNA5 async global->LDS copy (ASYNCcnt-tracked, no VGPR round trip).
// LDS offset = low 32 bits of the flat pointer (aperture occupies bits 63:32).
__device__ __forceinline__ void async_copy_b128(void* lds, const void* gaddr) {
  unsigned lo = (unsigned)(uintptr_t)lds;
  asm volatile("global_load_async_to_lds_b128 %0, %1, off"
               :: "v"(lo), "v"(gaddr) : "memory");
}
#define WAIT_ASYNCCNT(n) asm volatile("s_wait_asynccnt " #n ::: "memory")

// ---------- CDNA5 Tensor Data Mover: one descriptor moves a 128x32 f16 tile ----
// D# per CDNA5 ISA:
//  group0: [1:0]=count=1, [63:32]=lds_addr, [120:64]=global_addr, [127:126]=type=2
//  group1: data_size=1 (2B), pad_enable=1, pad_interval=3 (16 DW = 64B rows),
//          pad_amount=3 (4 DW = 16B pad -> LDS row stride 80B),
//          tensor_dim0=K, tensor_dim1=128, tile_dim0=32, tile_dim1=128,
//          tensor_dim0_stride=K
__device__ __forceinline__ void tdm_load_tile(void* lds, const _Float16* gsrc, int K) {
  unsigned lo = (unsigned)(uintptr_t)lds;
  unsigned long long ga = (unsigned long long)(uintptr_t)gsrc;
  unsigned td0 = (unsigned)K;

  v4u g0;
  g0[0] = 1u;                                   // count=1 (valid user descriptor)
  g0[1] = lo;                                   // lds_addr
  g0[2] = (unsigned)ga;                         // global_addr[31:0]
  g0[3] = (unsigned)(ga >> 32) | (2u << 30);    // global_addr[56:32] | type=2

  v8i g1;
  g1[0] = (int)((1u << 16) | (1u << 20) | (3u << 22) | (3u << 25));
  g1[1] = (int)((td0 & 0xFFFFu) << 16);                       // tensor_dim0[15:0]
  g1[2] = (int)(((td0 >> 16) & 0xFFFFu) | (128u << 16));      // dim0[31:16] | tensor_dim1=128
  g1[3] = (int)(32u << 16);                                   // tile_dim0 = 32 elems
  g1[4] = (int)128u;                                          // tile_dim1 = 128, tile_dim2 = 0
  g1[5] = (int)td0;                                           // tensor_dim0_stride[31:0] = K
  g1[6] = 0;                                                  // stride hi, dim1_stride lo
  g1[7] = 0;

  v4i z;  z[0] = 0; z[1] = 0; z[2] = 0; z[3] = 0;
#if __clang_major__ >= 23
  v8i z8;
#pragma unroll
  for (int i = 0; i < 8; ++i) z8[i] = 0;
  __builtin_amdgcn_tensor_load_to_lds(g0, g1, z, z, z8, 0);
#else
  __builtin_amdgcn_tensor_load_to_lds(g0, g1, z, z, 0);
#endif
}

// ---------- fp32 -> fp16 convert (vectorized x4) ----------
__global__ __launch_bounds__(256) void cvt_f32_f16(const float* __restrict__ in,
                                                   _Float16* __restrict__ out,
                                                   int n4) {
  int i = blockIdx.x * 256 + threadIdx.x;
  if (i < n4) {
    float4 f = ((const float4*)in)[i];
    v4h h;
    h[0] = (_Float16)f.x; h[1] = (_Float16)f.y;
    h[2] = (_Float16)f.z; h[3] = (_Float16)f.w;
    ((v4h*)out)[i] = h;
  }
}

// ---------- projection GEMM: Y[M,N] = X[M,K] * W[N,K]^T + bias ----------
// block: 256 threads (8 waves), tile M=128 (16 rows/wave), N=128, K-step 32.
// Double-buffered LDS fed by the Tensor Data Mover (wave 0 issues, TENSORcnt).
#define GBM 128
#define GBN 128
#define GBK 32
#define LPAD 8                      // LDS row stride 40 halfs (80B) == TDM pad
#define GROWS (GBK + LPAD)

__global__ __launch_bounds__(256) void proj_gemm(const _Float16* __restrict__ X,
                                                 const _Float16* __restrict__ W,
                                                 const float* __restrict__ bias,
                                                 _Float16* __restrict__ Y,
                                                 int M, int N, int K) {
  __shared__ _Float16 As[2][GBM][GROWS];   // 2 x 10240 B
  __shared__ _Float16 Bs[2][GBN][GROWS];   // 2 x 10240 B

  const int tid  = threadIdx.x;
  const int wave = tid >> 5;
  const int lane = tid & 31;
  const int m0   = blockIdx.y * GBM;
  const int n0   = blockIdx.x * GBN;

  v8f acc[8];
#pragma unroll
  for (int t = 0; t < 8; ++t) acc[t] = zero_v8f();

  const int base = (lane < 16) ? 0 : 8;    // A-frag K base
  const int koff = (lane < 16) ? 0 : 8;    // B-frag K base
  const int frow = wave * 16 + (lane & 15);

  const _Float16* Xb = X + (size_t)m0 * K;
  const _Float16* Wb = W + (size_t)n0 * K;

  // prologue: TDM-stage tile 0 into buffer 0 (wave 0 only; TDM ignores EXEC)
  if (wave == 0) {
    tdm_load_tile(&As[0][0][0], Xb, K);
    tdm_load_tile(&Bs[0][0][0], Wb, K);
  }

  int ib = 0;
  for (int k0 = 0; k0 < K; k0 += GBK, ib ^= 1) {
    if (wave == 0) {
      if (k0 + GBK < K) {
        tdm_load_tile(&As[ib ^ 1][0][0], Xb + k0 + GBK, K);
        tdm_load_tile(&Bs[ib ^ 1][0][0], Wb + k0 + GBK, K);
        // same-wave tensor ops complete in order: <=2 outstanding means
        // the two transfers for tile k0 have landed.
        __builtin_amdgcn_s_wait_tensorcnt(2);
      } else {
        __builtin_amdgcn_s_wait_tensorcnt(0);
      }
    }
    __syncthreads();

    // A fragment for this wave's 16 rows
    v8h alo = *(const v8h*)&As[ib][frow][base];
    v8h ahi = *(const v8h*)&As[ib][frow][base + 16];
    v16h a  = cat8(alo, ahi);

#pragma unroll
    for (int t = 0; t < 8; ++t) {
      int bn = t * 16 + (lane & 15);
      v8h b0 = *(const v8h*)&Bs[ib][bn][koff];
      v8h b1 = *(const v8h*)&Bs[ib][bn][koff + 16];
      v16h b = cat8(b0, b1);
      acc[t] = __builtin_amdgcn_wmma_f32_16x16x32_f16(
          false, a, false, b, (short)0, acc[t], false, false);
    }
    __syncthreads();   // all reads of buf[ib] done before it is refilled
  }

  // store C: row = m0 + wave*16 + v + 8*(lane>=16), col = n0 + t*16 + lane%16
  const int rbase = m0 + wave * 16 + ((lane < 16) ? 0 : 8);
#pragma unroll
  for (int t = 0; t < 8; ++t) {
    int c = n0 + t * 16 + (lane & 15);
    float b = bias[c];
#pragma unroll
    for (int v = 0; v < 8; ++v) {
      Y[(size_t)(rbase + v) * N + c] = (_Float16)(acc[t][v] + b);
    }
  }
}

// ---------- GQA flash attention ----------
// grid: (T/128, 32 heads); block 256 threads = 8 waves, each wave owns 16 query rows.
#define ABQ 128
#define ABK 32

__global__ __launch_bounds__(256) void gqa_attn(const _Float16* __restrict__ Q,  // [T, 4096]
                                                const _Float16* __restrict__ Km, // [T, 2048]
                                                const _Float16* __restrict__ Vm, // [T, 2048]
                                                float* __restrict__ O) {         // [T, 4096]
  __shared__ _Float16 Ks[ABK][HD + LPAD];        // K tile, row-major  (32 x 136)
  __shared__ _Float16 Vs[HD][ABK + LPAD];        // V tile, transposed (128 x 40)
  __shared__ _Float16 Ps[8][16][ABK + LPAD];     // per-wave P tile    (16 x 40)

  const int tid  = threadIdx.x;
  const int wave = tid >> 5;
  const int lane = tid & 31;
  const int h    = blockIdx.y;        // query head
  const int kv   = h & (NKV - 1);     // kv head = h % 16 (h = g*NKV + kv)
  const int qb0  = blockIdx.x * ABQ;

  const int base  = (lane < 16) ? 0 : 8;   // A-frag K base
  const int koff  = (lane < 16) ? 0 : 8;   // B-frag K base
  const int hiAdd = (lane < 16) ? 0 : 8;   // C-layout row offset
  const int qRowA = qb0 + wave * 16 + (lane & 15);  // A-frag row for this lane
  const int myQEnd = qb0 + wave * 16 + 15;          // max query row in wave

  // load Q fragments (16 rows x 128 d = 4 A-frags), straight from global
  v16h qf[4];
  {
    const _Float16* qp = Q + (size_t)qRowA * C_DIM + h * HD;
#pragma unroll
    for (int c = 0; c < 4; ++c) {
      v8h lo = *(const v8h*)(qp + c * 32 + base);
      v8h hi = *(const v8h*)(qp + c * 32 + base + 16);
      qf[c] = cat8(lo, hi);
    }
  }

  v8f acc_o[8];
#pragma unroll
  for (int t = 0; t < 8; ++t) acc_o[t] = zero_v8f();
  float mrow[8], lrow[8];
#pragma unroll
  for (int v = 0; v < 8; ++v) { mrow[v] = -__builtin_inff(); lrow[v] = 0.0f; }

  const int nsteps = (qb0 + ABQ) / ABK;   // all keys this block can see (causal)
  for (int s = 0; s < nsteps; ++s) {
    const int kBase = s * ABK;

    // ---- cooperative stage of K (row-major, async) and V (transposed) ----
    {
      int r = tid >> 3;            // key 0..31
      int c = (tid & 7) * 16;      // d 0..112
      const _Float16* ksrc = Km + (size_t)(kBase + r) * KVD + kv * HD + c;
      async_copy_b128(&Ks[r][c],     ksrc);
      async_copy_b128(&Ks[r][c + 8], ksrc + 8);

      const _Float16* vsrc = Vm + (size_t)(kBase + r) * KVD + kv * HD + c;
      v8h t0 = *(const v8h*)vsrc;
      v8h t1 = *(const v8h*)(vsrc + 8);
#pragma unroll
      for (int j = 0; j < 8; ++j) {
        Vs[c + j][r]     = t0[j];
        Vs[c + 8 + j][r] = t1[j];
      }
      WAIT_ASYNCCNT(0);
    }
    __syncthreads();

    if (kBase <= myQEnd) {   // wave-uniform predicate: EXEC stays all-ones
      // ---- S = Q K^T, two 16-key column tiles ----
      v8f sacc[2];
#pragma unroll
      for (int c2 = 0; c2 < 2; ++c2) {
        sacc[c2] = zero_v8f();
        int krow = c2 * 16 + (lane & 15);
#pragma unroll
        for (int dc = 0; dc < 4; ++dc) {
          v8h b0 = *(const v8h*)&Ks[krow][dc * 32 + koff];
          v8h b1 = *(const v8h*)&Ks[krow][dc * 32 + koff + 16];
          v16h b = cat8(b0, b1);
          sacc[c2] = __builtin_amdgcn_wmma_f32_16x16x32_f16(
              false, qf[dc], false, b, (short)0, sacc[c2], false, false);
        }
      }

      // ---- masked online softmax over this 16x32 stripe ----
      float fexp[8];
      const int kc0 = kBase + (lane & 15);
      const int kc1 = kc0 + 16;
#pragma unroll
      for (int v = 0; v < 8; ++v) {
        int qr = qb0 + wave * 16 + v + hiAdd;
        float s0 = (kc0 <= qr) ? sacc[0][v] * SCALE_F : -__builtin_inff();
        float s1 = (kc1 <= qr) ? sacc[1][v] * SCALE_F : -__builtin_inff();
        float rm = fmaxf(s0, s1);
#pragma unroll
        for (int off = 1; off < 16; off <<= 1)
          rm = fmaxf(rm, __shfl_xor(rm, off, 16));
        float mnew = fmaxf(mrow[v], rm);
        float fe, p0, p1;
        if (mnew == -__builtin_inff()) { fe = 1.0f; p0 = 0.0f; p1 = 0.0f; }
        else {
          fe = __expf(mrow[v] - mnew);
          p0 = __expf(s0 - mnew);
          p1 = __expf(s1 - mnew);
        }
        float rs = p0 + p1;
#pragma unroll
        for (int off = 1; off < 16; off <<= 1)
          rs += __shfl_xor(rs, off, 16);
        lrow[v] = lrow[v] * fe + rs;
        mrow[v] = mnew;
        fexp[v] = fe;
        // scatter P into per-wave LDS tile (C-layout -> row-major)
        Ps[wave][v + hiAdd][lane & 15]        = (_Float16)p0;
        Ps[wave][v + hiAdd][16 + (lane & 15)] = (_Float16)p1;
      }
      // rescale running output
#pragma unroll
      for (int t = 0; t < 8; ++t)
#pragma unroll
        for (int v = 0; v < 8; ++v) acc_o[t][v] *= fexp[v];

      // ---- P fragment (A: 16x32) from LDS ----
      {
        int prow = lane & 15;
        v8h plo = *(const v8h*)&Ps[wave][prow][base];
        v8h phi = *(const v8h*)&Ps[wave][prow][base + 16];
        v16h pf = cat8(plo, phi);

        // ---- O += P * V over 8 d-tiles ----
#pragma unroll
        for (int t = 0; t < 8; ++t) {
          int d = t * 16 + (lane & 15);
          v8h v0 = *(const v8h*)&Vs[d][koff];
          v8h v1 = *(const v8h*)&Vs[d][koff + 16];
          v16h vf = cat8(v0, v1);
          acc_o[t] = __builtin_amdgcn_wmma_f32_16x16x32_f16(
              false, pf, false, vf, (short)0, acc_o[t], false, false);
        }
      }
    }
    __syncthreads();
  }

  // ---- normalize and store (fp32 output) ----
#pragma unroll
  for (int v = 0; v < 8; ++v) {
    int qr = qb0 + wave * 16 + v + hiAdd;
    float inv = 1.0f / lrow[v];
#pragma unroll
    for (int t = 0; t < 8; ++t) {
      int c = h * HD + t * 16 + (lane & 15);
      O[(size_t)qr * C_DIM + c] = acc_o[t][v] * inv;
    }
  }
}

// ---------- host launcher ----------
extern "C" void kernel_launch(void* const* d_in, const int* in_sizes, int n_in,
                              void* d_out, int out_size, void* d_ws, size_t ws_size,
                              hipStream_t stream) {
  (void)in_sizes; (void)n_in; (void)out_size; (void)ws_size;
  const float* x  = (const float*)d_in[0];
  const float* Wq = (const float*)d_in[1];
  const float* bq = (const float*)d_in[2];
  const float* Wk = (const float*)d_in[3];
  const float* bk = (const float*)d_in[4];
  const float* Wv = (const float*)d_in[5];
  const float* bv = (const float*)d_in[6];
  float* out = (float*)d_out;

  char* ws = (char*)d_ws;
  _Float16* xb  = (_Float16*)ws;  ws += (size_t)T_SEQ * C_DIM * 2;   // 16 MB
  _Float16* Wqb = (_Float16*)ws;  ws += (size_t)C_DIM * C_DIM * 2;   // 32 MB
  _Float16* Wkb = (_Float16*)ws;  ws += (size_t)KVD   * C_DIM * 2;   // 16 MB
  _Float16* Wvb = (_Float16*)ws;  ws += (size_t)KVD   * C_DIM * 2;   // 16 MB
  _Float16* qb  = (_Float16*)ws;  ws += (size_t)T_SEQ * C_DIM * 2;   // 16 MB
  _Float16* kb  = (_Float16*)ws;  ws += (size_t)T_SEQ * KVD   * 2;   //  8 MB
  _Float16* vb  = (_Float16*)ws;  ws += (size_t)T_SEQ * KVD   * 2;   //  8 MB

  auto cvt = [&](const float* src, _Float16* dst, size_t n) {
    int n4 = (int)(n / 4);
    cvt_f32_f16<<<(n4 + 255) / 256, 256, 0, stream>>>(src, dst, n4);
  };
  cvt(x,  xb,  (size_t)T_SEQ * C_DIM);
  cvt(Wq, Wqb, (size_t)C_DIM * C_DIM);
  cvt(Wk, Wkb, (size_t)KVD * C_DIM);
  cvt(Wv, Wvb, (size_t)KVD * C_DIM);

  // projections: y = x W^T + b   (f16 outputs)
  proj_gemm<<<dim3(C_DIM / GBN, T_SEQ / GBM), 256, 0, stream>>>(
      xb, Wqb, bq, qb, T_SEQ, C_DIM, C_DIM);
  proj_gemm<<<dim3(KVD / GBN, T_SEQ / GBM), 256, 0, stream>>>(
      xb, Wkb, bk, kb, T_SEQ, KVD, C_DIM);
  proj_gemm<<<dim3(KVD / GBN, T_SEQ / GBM), 256, 0, stream>>>(
      xb, Wvb, bv, vb, T_SEQ, KVD, C_DIM);

  // fused causal GQA attention
  gqa_attn<<<dim3(T_SEQ / ABQ, NHEADS), 256, 0, stream>>>(qb, kb, vb, out);
}
